// DVAE_PYG_11897059410770
// MI455X (gfx1250) — matrix operand order repaired and checked by reading, not compiled, not run on hardware
//
#include <hip/hip_runtime.h>
#include <hip/hip_bf16.h>

// ---------------------------------------------------------------------------
// D-VAE GNN encoder for MI455X (gfx1250, wave32, WMMA f32_16x16x32_f16)
//
//  * ids-concat part of Hcat is just a per-node bias column of Wg/Wm.
//  * H rows are write-once => gated message G[b,u,:] computed once per node
//    and cached in workspace; Hmsg is a short sparse weighted sum over G.
//  * Fully batch-parallel: each block owns 16 batch rows for all 32 steps.
//  * A-fragments via ds_load_b128 pairs; B-fragments via global_load_b128.
// ---------------------------------------------------------------------------

#define B_   512
#define N_   32
#define NVT_ 32
#define HS_  501
#define VS_  533      // HS + N
#define KP_  512      // padded HS (K dim, multiple of 32)
#define GH_  1536     // padded 3*HS (gru gate dim, multiple of 16)
#define BM_  16       // batch rows per block (WMMA M)

typedef __attribute__((ext_vector_type(16))) _Float16 v16h;
typedef __attribute__((ext_vector_type(8)))  float    v8f;
typedef __attribute__((ext_vector_type(8)))  unsigned int v8u;

// A-matrix fragment (16x32 f16, row-major, contiguous k order, same as load_frag_b):
// lane L: row m = L&15, kh = (L>>4)*16 ; VGPR0-3 = K kh..kh+7, VGPR4-7 = K kh+8..kh+15
// Both halves are 16B-aligned -> two b128 loads.
__device__ __forceinline__ v16h load_frag_a(const _Float16* base) {
  const uint4* p = (const uint4*)base;   // uint4 index 0 = words 0..3, index 1 = words 4..7
  uint4 lo = p[0];
  uint4 hi = p[1];
  v8u u;
  u[0] = lo.x; u[1] = lo.y; u[2] = lo.z; u[3] = lo.w;
  u[4] = hi.x; u[5] = hi.y; u[6] = hi.z; u[7] = hi.w;
  return __builtin_bit_cast(v16h, u);
}

// B-matrix fragment (32x16 f16): lane L: col n = L&15, K-run = (L>>4)*16 .. +15
// Weights stored as W[n][k] row-major -> one contiguous 32B load (2x b128).
__device__ __forceinline__ v16h load_frag_b(const _Float16* base) {
  return *(const v16h*)base;
}

__device__ __forceinline__ float sigm(float x) { return 1.0f / (1.0f + __expf(-x)); }

// ---------------------------------------------------------------------------
// Prep: convert f32 weights to zero-padded f16 GEMM-B layouts.
// ---------------------------------------------------------------------------
__global__ void dvae_prep(const float* __restrict__ Wg, const float* __restrict__ Wm,
                          const float* __restrict__ Whh,
                          _Float16* __restrict__ WgB, _Float16* __restrict__ WmB,
                          _Float16* __restrict__ WhhB) {
  int idx = blockIdx.x * blockDim.x + threadIdx.x;
  if (idx < KP_ * KP_) {
    int n = idx >> 9, k = idx & (KP_ - 1);
    float g = 0.f, m = 0.f;
    if (n < HS_ && k < HS_) { g = Wg[n * VS_ + k]; m = Wm[n * VS_ + k]; }
    WgB[idx] = (_Float16)g;
    WmB[idx] = (_Float16)m;
  }
  if (idx < GH_ * KP_) {
    int n = idx >> 9, k = idx & (KP_ - 1);
    float w = 0.f;
    if (n < 3 * HS_ && k < HS_) w = Whh[n * HS_ + k];
    WhhB[idx] = (_Float16)w;
  }
}

// ---------------------------------------------------------------------------
// Main persistent kernel: 32 blocks x 256 threads (8 waves). Each block owns
// batch rows [16*blk, 16*blk+16) and runs the whole 32-node scan.
// ---------------------------------------------------------------------------
__global__ __launch_bounds__(256, 1)
void dvae_main(const int*   __restrict__ node_types,
               const float* __restrict__ adj,
               const float* __restrict__ Wg,   const float* __restrict__ bg,
               const float* __restrict__ Wm,
               const float* __restrict__ W_ih, const float* __restrict__ b_ih,
               const float* __restrict__ b_hh,
               const _Float16* __restrict__ WgB,
               const _Float16* __restrict__ WmB,
               const _Float16* __restrict__ WhhB,
               float* __restrict__ G,           // [32 blk][32 node][16 m][512 h]
               float* __restrict__ Hg)          // [512 b][512 h]
{
  extern __shared__ char smem[];
  float*    ghF   = (float*)smem;                       // 16*1536 f32  (gru pre-act)
  float*    hmsgF = ghF + BM_ * GH_;                    // 16*512  f32
  float*    adjC  = hmsgF + BM_ * KP_;                  // 16*32   f32
  int*      typC  = (int*)(adjC + BM_ * N_);            // 16      i32
  _Float16* hmsgH = (_Float16*)(typC + BM_);            // 16*512  f16
  _Float16* hvH   = hmsgH + BM_ * KP_;                  // 16*512  f16

  const int tid  = threadIdx.x;
  const int wave = tid >> 5;
  const int lane = tid & 31;
  const int blk  = blockIdx.x;
  const int b0   = blk * BM_;
  float* Gblk = G + (size_t)blk * (N_ * BM_ * KP_);

  const int mrow  = lane & 15;
  const int kh    = (lane >> 4) * 16;   // A-frag K run
  const int ksec  = (lane >> 4) * 16;   // B-frag K run
  const int mBase = (lane >> 4) * 8;    // D-frag row base

  for (int v = 0; v < N_; ++v) {
    // ---- stage adjacency column adj[b, u, v] and node types ---------------
    if (tid < BM_ * N_) {
      int m = tid >> 5, u = tid & 31;
      adjC[tid] = adj[(size_t)(b0 + m) * (N_ * N_) + u * N_ + v];
    }
    if (tid < BM_) typC[tid] = node_types[(b0 + tid) * N_ + v];
    __syncthreads();

    // ---- Hmsg[m,h] = sum_{u<v} adj * G[u][m][h] ---------------------------
    for (int idx = tid; idx < BM_ * KP_; idx += 256) {
      int m = idx >> 9, h = idx & (KP_ - 1);
      float acc = 0.f;
      const float* gu = Gblk + m * KP_ + h;
      for (int u = 0; u < v; ++u) {
        float a = adjC[m * N_ + u];
        if (a != 0.f) acc += a * gu[(size_t)u * (BM_ * KP_)];
      }
      hmsgF[idx] = acc;
      hmsgH[idx] = (_Float16)acc;
    }
    __syncthreads();

    // ---- gh = Hmsg @ Whh^T : WMMA 16 x 1536 x 512, 96 n-tiles, 12/wave ----
    for (int tt = 0; tt < 12; ++tt) {
      const int t = wave * 12 + tt;
      const _Float16* bbase = WhhB + (size_t)(t * 16 + mrow) * KP_ + ksec;
      const _Float16* abase = hmsgH + mrow * KP_ + kh;
      __builtin_prefetch(bbase, 0, 3);
      v8f acc = {};
#pragma unroll 4
      for (int kt = 0; kt < KP_ / 32; ++kt) {
        v16h a = load_frag_a(abase + kt * 32);
        v16h b = load_frag_b(bbase + kt * 32);
        acc = __builtin_amdgcn_wmma_f32_16x16x32_f16(false, a, false, b,
                                                     (short)0, acc, false, false);
      }
      const int col = t * 16 + mrow;
#pragma unroll
      for (int r = 0; r < 8; ++r) ghF[(mBase + r) * GH_ + col] = acc[r];
    }
    __syncthreads();

    // ---- GRU elementwise: Hv = (1-z)*n + z*Hmsg ---------------------------
    for (int idx = tid; idx < BM_ * KP_; idx += 256) {
      int m = idx >> 9, h = idx & (KP_ - 1);
      float hv = 0.f;
      if (h < HS_) {
        int ty = typC[m];
        float gh_r = ghF[m * GH_ + h]            + b_hh[h];
        float gh_z = ghF[m * GH_ + HS_ + h]      + b_hh[HS_ + h];
        float gh_n = ghF[m * GH_ + 2 * HS_ + h]  + b_hh[2 * HS_ + h];
        float gi_r = W_ih[(size_t)h * NVT_ + ty]             + b_ih[h];
        float gi_z = W_ih[(size_t)(HS_ + h) * NVT_ + ty]     + b_ih[HS_ + h];
        float gi_n = W_ih[(size_t)(2 * HS_ + h) * NVT_ + ty] + b_ih[2 * HS_ + h];
        float r = sigm(gi_r + gh_r);
        float z = sigm(gi_z + gh_z);
        float n = tanhf(gi_n + r * gh_n);
        hv = (1.0f - z) * n + z * hmsgF[idx];
      }
      hvH[idx] = (_Float16)hv;
      if (v == N_ - 1) Hg[(size_t)(b0 + m) * KP_ + h] = hv;
    }
    __syncthreads();

    // ---- gated message for node v: G = sigm(Hv@Wg^T + bias)*(Hv@Wm^T + bias)
    // 32 h-tiles, 4 per wave; the two accumulators share each A fragment.
    for (int tt = 0; tt < 4; ++tt) {
      const int t = wave * 4 + tt;
      const _Float16* gb = WgB + (size_t)(t * 16 + mrow) * KP_ + ksec;
      const _Float16* mb = WmB + (size_t)(t * 16 + mrow) * KP_ + ksec;
      const _Float16* abase = hvH + mrow * KP_ + kh;
      __builtin_prefetch(gb, 0, 3);
      __builtin_prefetch(mb, 0, 3);
      v8f accG = {};
      v8f accM = {};
#pragma unroll 4
      for (int kt = 0; kt < KP_ / 32; ++kt) {
        v16h a  = load_frag_a(abase + kt * 32);
        v16h bG = load_frag_b(gb + kt * 32);
        v16h bM = load_frag_b(mb + kt * 32);
        accG = __builtin_amdgcn_wmma_f32_16x16x32_f16(false, a, false, bG,
                                                      (short)0, accG, false, false);
        accM = __builtin_amdgcn_wmma_f32_16x16x32_f16(false, a, false, bM,
                                                      (short)0, accM, false, false);
      }
      const int h = t * 16 + mrow;
      float biasg = 0.f, biasm = 0.f;
      if (h < HS_) {
        biasg = Wg[(size_t)h * VS_ + HS_ + v] + bg[h];   // ids column + bias
        biasm = Wm[(size_t)h * VS_ + HS_ + v];
      }
#pragma unroll
      for (int r = 0; r < 8; ++r) {
        float g = 0.f;
        if (h < HS_) g = sigm(accG[r] + biasg) * (accM[r] + biasm);
        Gblk[((size_t)v * BM_ + (mBase + r)) * KP_ + h] = g;
      }
    }
    __threadfence_block();
    __syncthreads();
  }
}

// ---------------------------------------------------------------------------
// Regression head: mu[b,n] = Hg[b,:] . W_reg[n,:] + b_reg[n]
// ---------------------------------------------------------------------------
__global__ void dvae_mu(const float* __restrict__ Hg,
                        const float* __restrict__ W_reg,
                        const float* __restrict__ b_reg,
                        float* __restrict__ out) {
  int idx = blockIdx.x * blockDim.x + threadIdx.x;
  if (idx >= B_ * N_) return;
  int b = idx >> 5, n = idx & 31;
  const float* hrow = Hg + (size_t)b * KP_;
  const float* wrow = W_reg + (size_t)n * HS_;
  float acc = b_reg[n];
  for (int h = 0; h < HS_; ++h) acc += hrow[h] * wrow[h];
  out[idx] = acc;
}

// ---------------------------------------------------------------------------
extern "C" void kernel_launch(void* const* d_in, const int* in_sizes, int n_in,
                              void* d_out, int out_size, void* d_ws, size_t ws_size,
                              hipStream_t stream) {
  const int*   node_types = (const int*)  d_in[0];
  const float* adj   = (const float*)d_in[1];
  const float* Wg    = (const float*)d_in[2];
  const float* bg    = (const float*)d_in[3];
  const float* Wm    = (const float*)d_in[4];
  const float* W_ih  = (const float*)d_in[5];
  const float* b_ih  = (const float*)d_in[6];
  const float* W_hh  = (const float*)d_in[7];
  const float* b_hh  = (const float*)d_in[8];
  const float* W_reg = (const float*)d_in[9];
  const float* b_reg = (const float*)d_in[10];

  char* ws = (char*)d_ws;
  _Float16* WgB  = (_Float16*)(ws);                       // 512*512*2 = 512 KB
  _Float16* WmB  = (_Float16*)(ws + 524288);              // 512 KB
  _Float16* WhhB = (_Float16*)(ws + 1048576);             // 1536*512*2 = 1.5 MB
  float*    Hg   = (float*)   (ws + 2621440);             // 512*512*4 = 1 MB
  float*    G    = (float*)   (ws + 3670016);             // 32*32*16*512*4 = 32 MB

  // 1) weight conversion / padding
  dvae_prep<<<(GH_ * KP_ + 255) / 256, 256, 0, stream>>>(Wg, Wm, W_hh, WgB, WmB, WhhB);

  // 2) persistent 32-step scan, 32 blocks x 8 waves, dynamic LDS partition
  size_t smem = (size_t)(BM_ * GH_ + BM_ * KP_ + BM_ * N_ + BM_) * sizeof(float)
              + (size_t)(2 * BM_ * KP_) * sizeof(_Float16);   // ~162 KB
  dvae_main<<<32, 256, smem, stream>>>(node_types, adj, Wg, bg, Wm, W_ih, b_ih,
                                       b_hh, WgB, WmB, WhhB, G, Hg);

  // 3) regression head
  dvae_mu<<<(B_ * N_ + 255) / 256, 256, 0, stream>>>(Hg, W_reg, b_reg, (float*)d_out);
}
